// InterGraphMatchingLayer_87514253624203
// MI455X (gfx1250) — matrix operand order repaired and checked by reading, not compiled, not run on hardware
//
#include <hip/hip_runtime.h>
#include <hip/hip_bf16.h>
#include <math.h>

// ---------------------------------------------------------------------------
// InterGraphMatchingLayer on gfx1250 (MI455X).
//  * N^2*D GEMMs via v_wmma_f32_16x16x32_bf16 with split-bf16 (hi+lo) operands
//    (~fp32 accuracy at bf16-WMMA rate), fused flash-style: the 8192x8192
//    score matrices never touch HBM.
//  * Operands pre-split to bf16 hi/lo in global memory once; Y tiles are
//    double-buffered into LDS with global_load_async_to_lds_b128 (ASYNCcnt),
//    overlapping the L2 fetch with the WMMA stream.
// ---------------------------------------------------------------------------

typedef __attribute__((ext_vector_type(16))) __bf16 bf16x16;
typedef __attribute__((ext_vector_type(8)))  float  floatx8;

#define DF 128           // feature dim
#define BROWS 64         // rows of X per block
#define NWAVE 4          // waves per block (wave32)
#define JT 64            // j-columns per LDS tile
#define NTHREADS 128
#define LDSTRIDE (DF + 8)   // 136 bf16 = 272B row stride (16B multiple)

__device__ __forceinline__ floatx8 wmma_bf16(bf16x16 a, bf16x16 b, floatx8 c) {
  return __builtin_amdgcn_wmma_f32_16x16x32_bf16(
      /*neg_a=*/false, a, /*neg_b=*/false, b,
      /*c_mod=*/(short)0, c, /*reuse_a=*/false, /*reuse_b=*/false);
}

// K position inside a 16x32 bf16 A fragment: VGPR pair v, lane half.
__device__ __forceinline__ int a_kpat(int v, int half) {
  return 2 * v + 16 * half;
}

// Async copy of one 16B chunk: global -> LDS, tracked by ASYNCcnt.
// (flat low 32 bits of a __shared__ pointer are the LDS byte offset)
__device__ __forceinline__ void async_copy_b128(const void* gptr, void* lptr) {
  const unsigned lds = (unsigned)(uintptr_t)lptr;
  asm volatile("global_load_async_to_lds_b128 %0, %1, off"
               :: "v"(lds), "v"(gptr) : "memory");
}
__device__ __forceinline__ void wait_async0() {
  asm volatile("s_wait_asynccnt 0x0" ::: "memory");
}

// ---------------------------------------------------------------------------
// Elementwise fp32 -> bf16 hi/lo split (run once per operand matrix).
// ---------------------------------------------------------------------------
__global__ void __launch_bounds__(256)
split_bf16(const float* __restrict__ in, __bf16* __restrict__ hi,
           __bf16* __restrict__ lo, int n) {
  const int i = blockIdx.x * 256 + threadIdx.x;
  if (i < n) {
    float v = in[i];
    __bf16 h = (__bf16)v;
    hi[i] = h;
    lo[i] = (__bf16)(v - (float)h);
  }
}

// ---------------------------------------------------------------------------
// Fused: acc[i,:] = sum_j S_ij * Y[j,:]   with S = X @ Y^T (optionally
// zero-diagonal), and rs[i] = sum_j |S_ij|.  X/Y given as bf16 hi/lo pairs.
// ---------------------------------------------------------------------------
__global__ void __launch_bounds__(NTHREADS)
fused_sim_acc(const __bf16* __restrict__ Xhi, const __bf16* __restrict__ Xlo,
              const __bf16* __restrict__ Yhi, const __bf16* __restrict__ Ylo,
              float* __restrict__ acc_out, float* __restrict__ rs_out,
              int n_total, int zero_diag)
{
  __shared__ __bf16 sYhi[2][JT][LDSTRIDE];
  __shared__ __bf16 sYlo[2][JT][LDSTRIDE];
  __shared__ __bf16 sShi[NWAVE][16][JT + 8];
  __shared__ __bf16 sSlo[NWAVE][16][JT + 8];

  const int tid  = threadIdx.x;
  const int w    = tid >> 5;
  const int lane = tid & 31;
  const int half = lane >> 4;
  const int l16  = lane & 15;
  const int rowBase = blockIdx.x * BROWS;
  const int aRow = rowBase + w * 16 + l16;    // A-fragment row (M = lane%16)

  // --- X stripe A fragments (bf16 hi/lo), resident for the whole block ---
  bf16x16 Ahi[4], Alo[4];
#pragma unroll
  for (int kt = 0; kt < 4; ++kt) {
#pragma unroll
    for (int v = 0; v < 8; ++v) {
      const int k = kt * 32 + a_kpat(v, half);
      Ahi[kt][2 * v]     = Xhi[(size_t)aRow * DF + k];
      Ahi[kt][2 * v + 1] = Xhi[(size_t)aRow * DF + k + 1];
      Alo[kt][2 * v]     = Xlo[(size_t)aRow * DF + k];
      Alo[kt][2 * v + 1] = Xlo[(size_t)aRow * DF + k + 1];
    }
  }

  floatx8 acc[8];
#pragma unroll
  for (int i = 0; i < 8; ++i) acc[i] = {};
  floatx8 rs = {};

  const int ntiles = n_total / JT;

  // Issue async fill of buffer 0 (each lane copies 16 x 16B chunks:
  // {hi,lo} x 64 rows x 16 chunks = 2048 chunks / 128 lanes).
  auto issue_tile = [&](int jb, int buf) {
#pragma unroll
    for (int q = 0; q < 16; ++q) {
      const int idx = q * NTHREADS + tid;        // 0..2047
      const int arr = idx >> 10;                 // 0 = hi, 1 = lo
      const int rem = idx & 1023;
      const int r = rem >> 4, c = rem & 15;      // row, 16B chunk
      const __bf16* g = (arr ? Ylo : Yhi) + (size_t)(jb + r) * DF + c * 8;
      void* l = arr ? (void*)&sYlo[buf][r][c * 8] : (void*)&sYhi[buf][r][c * 8];
      async_copy_b128(g, l);
    }
  };

  issue_tile(0, 0);

  for (int t = 0; t < ntiles; ++t) {
    const int jb  = t * JT;
    const int cur = t & 1;
    wait_async0();        // this wave's copies for tile t have landed
    __syncthreads();      // all waves' copies landed; prev buffer fully read
    if (t + 1 < ntiles) issue_tile(jb + JT, cur ^ 1);

    // ---- stage 1: S stripe (16 x JT) = Xstripe @ Ytile^T ----
#pragma unroll
    for (int nt = 0; nt < JT / 16; ++nt) {
      floatx8 c = {};
      const int jl = nt * 16 + l16;              // B operand: N = lane%16 -> j
#pragma unroll
      for (int kt = 0; kt < 4; ++kt) {
        bf16x16 Bhi, Blo;
#pragma unroll
        for (int v = 0; v < 8; ++v) {
          const int k = kt * 32 + 16 * half + 2 * v;
          Bhi[2 * v]     = sYhi[cur][jl][k];
          Bhi[2 * v + 1] = sYhi[cur][jl][k + 1];
          Blo[2 * v]     = sYlo[cur][jl][k];
          Blo[2 * v + 1] = sYlo[cur][jl][k + 1];
        }
        c = wmma_bf16(Ahi[kt], Bhi, c);
        c = wmma_bf16(Ahi[kt], Blo, c);
        c = wmma_bf16(Alo[kt], Bhi, c);
      }
      // diag zero + |S| rowsum + stash S as bf16 hi/lo for stage 2
#pragma unroll
      for (int r = 0; r < 8; ++r) {
        const int m  = r + 8 * half;             // C-layout row
        const int gi = rowBase + w * 16 + m;
        const int gj = jb + nt * 16 + l16;
        float sv = c[r];
        if (zero_diag && gi == gj) sv = 0.0f;
        rs[r] += fabsf(sv);
        __bf16 h = (__bf16)sv;
        sShi[w][m][nt * 16 + l16] = h;
        sSlo[w][m][nt * 16 + l16] = (__bf16)(sv - (float)h);
      }
    }
    __builtin_amdgcn_wave_barrier();  // stripe LDS is wave-private; order ds ops

    // ---- stage 2: acc (16 x 128) += S (16 x JT) @ Ytile (JT x 128) ----
#pragma unroll
    for (int kt2 = 0; kt2 < JT / 32; ++kt2) {
      bf16x16 A2hi, A2lo;
#pragma unroll
      for (int v = 0; v < 8; ++v) {
        const int k = kt2 * 32 + a_kpat(v, half);
        A2hi[2 * v]     = sShi[w][l16][k];
        A2hi[2 * v + 1] = sShi[w][l16][k + 1];
        A2lo[2 * v]     = sSlo[w][l16][k];
        A2lo[2 * v + 1] = sSlo[w][l16][k + 1];
      }
#pragma unroll
      for (int nt2 = 0; nt2 < 8; ++nt2) {
        bf16x16 B2hi, B2lo;
        const int col = nt2 * 16 + l16;
#pragma unroll
        for (int v = 0; v < 8; ++v) {
          const int kk = kt2 * 32 + 16 * half + 2 * v;
          B2hi[2 * v]     = sYhi[cur][kk][col];
          B2hi[2 * v + 1] = sYhi[cur][kk + 1][col];
          B2lo[2 * v]     = sYlo[cur][kk][col];
          B2lo[2 * v + 1] = sYlo[cur][kk + 1][col];
        }
        acc[nt2] = wmma_bf16(A2hi, B2hi, acc[nt2]);
        acc[nt2] = wmma_bf16(A2hi, B2lo, acc[nt2]);
        acc[nt2] = wmma_bf16(A2lo, B2hi, acc[nt2]);
      }
    }
  }

  // ---- epilogue: rowsum reduce (within each 16-lane half) + stores ----
  if (zero_diag) {
#pragma unroll
    for (int m = 1; m < 16; m <<= 1)
#pragma unroll
      for (int r = 0; r < 8; ++r) rs[r] += __shfl_xor(rs[r], m, 32);
    if (l16 == 0) {
#pragma unroll
      for (int r = 0; r < 8; ++r)
        rs_out[rowBase + w * 16 + r + 8 * half] = rs[r];
    }
  }
#pragma unroll
  for (int nt = 0; nt < 8; ++nt)
#pragma unroll
    for (int r = 0; r < 8; ++r)
      acc_out[(size_t)(rowBase + w * 16 + r + 8 * half) * DF + nt * 16 + l16] =
          acc[nt][r];
}

// ---------------------------------------------------------------------------
// Per-row L1 normalization: out[i,:] = in[i,:] / max(sum|in[i,:]|, eps)
// ---------------------------------------------------------------------------
__global__ void __launch_bounds__(DF)
l1norm_rows(const float* __restrict__ in, float* __restrict__ out)
{
  __shared__ float red[4];
  const int row = blockIdx.x, col = threadIdx.x;
  float v = in[row * DF + col];
  float a = fabsf(v);
#pragma unroll
  for (int m = 1; m < 32; m <<= 1) a += __shfl_xor(a, m, 32);
  if ((col & 31) == 0) red[col >> 5] = a;
  __syncthreads();
  float s = red[0] + red[1] + red[2] + red[3];
  out[row * DF + col] = v / fmaxf(s, 1e-12f);
}

// ---------------------------------------------------------------------------
// Fused message/gate epilogue for one row:
//   m = Linear(acc*inv, Wd, bd);  s = Linear(fself, Ws, bs)
//   g = sigmoid(Linear(m,W1,b1) + Linear(s,W2,b2))
//   out = add + g*m + (1-g)*s
// Weights (4 x 64KB) stay L2-resident across the 8192 blocks.
// ---------------------------------------------------------------------------
__global__ void __launch_bounds__(DF)
gate_rows(const float* __restrict__ acc, const float* __restrict__ rs,
          const float* __restrict__ fself,
          const float* __restrict__ Wd, const float* __restrict__ bd,
          const float* __restrict__ Ws, const float* __restrict__ bs,
          const float* __restrict__ W1, const float* __restrict__ b1,
          const float* __restrict__ W2, const float* __restrict__ b2,
          const float* __restrict__ addv, float* __restrict__ out, int n_total)
{
  __shared__ float xrow[DF], frow[DF], mrow[DF], srow[DF];
  const int row = blockIdx.x, col = threadIdx.x;
  const float inv = 1.0f / (fmaxf(rs[row], 1e-12f) * (float)n_total);
  xrow[col] = acc[row * DF + col] * inv;
  frow[col] = fself[row * DF + col];
  __syncthreads();
  float mv = bd[col], sv = bs[col];
  for (int k = 0; k < DF; ++k) {
    mv = fmaf(xrow[k], Wd[col * DF + k], mv);
    sv = fmaf(frow[k], Ws[col * DF + k], sv);
  }
  mrow[col] = mv; srow[col] = sv;
  __syncthreads();
  float gp = b1[col] + b2[col];
  for (int k = 0; k < DF; ++k) {
    gp = fmaf(mrow[k], W1[col * DF + k], gp);
    gp = fmaf(srow[k], W2[col * DF + k], gp);
  }
  const float g = 1.0f / (1.0f + __expf(-gp));
  out[row * DF + col] = addv[row * DF + col] + g * mv + (1.0f - g) * sv;
}

// ---------------------------------------------------------------------------
extern "C" void kernel_launch(void* const* d_in, const int* in_sizes, int n_in,
                              void* d_out, int out_size, void* d_ws, size_t ws_size,
                              hipStream_t stream) {
  const float* f1   = (const float*)d_in[0];
  const float* f2   = (const float*)d_in[1];
  const float* Wd1  = (const float*)d_in[2];
  const float* bd1  = (const float*)d_in[3];
  const float* Wd2  = (const float*)d_in[4];
  const float* bd2  = (const float*)d_in[5];
  const float* Ws1  = (const float*)d_in[6];
  const float* bs1  = (const float*)d_in[7];
  const float* Ws2  = (const float*)d_in[8];
  const float* bs2  = (const float*)d_in[9];
  const float* g1W1 = (const float*)d_in[10];
  const float* g1b1 = (const float*)d_in[11];
  const float* g1W2 = (const float*)d_in[12];
  const float* g1b2 = (const float*)d_in[13];
  const float* g2W1 = (const float*)d_in[14];
  const float* g2b1 = (const float*)d_in[15];
  const float* g2W2 = (const float*)d_in[16];
  const float* g2b2 = (const float*)d_in[17];

  const int N = in_sizes[0] / DF;       // 8192
  const size_t ND = (size_t)N * DF;

  float* user_d1 = (float*)d_out;       // output 0
  float* user_d2 = user_d1 + ND;

  float* ws   = (float*)d_ws;
  float* acc1 = ws;  ws += ND;
  float* rs1  = ws;  ws += N;
  float* two1 = ws;  ws += ND;
  float* f1n  = ws;  ws += ND;
  float* acc2 = ws;  ws += ND;
  float* rs2  = ws;  ws += N;
  float* two2 = ws;  ws += ND;
  float* nrm2 = ws;  ws += ND;
  __bf16* bh = (__bf16*)ws;
  __bf16* f1h  = bh;  bh += ND;   // f1 hi/lo
  __bf16* f1l  = bh;  bh += ND;
  __bf16* f2h  = bh;  bh += ND;   // f2 hi/lo
  __bf16* f2l  = bh;  bh += ND;
  __bf16* f1nh = bh;  bh += ND;   // f1_new hi/lo
  __bf16* f1nl = bh;  bh += ND;

  const dim3 gblk(N / BROWS);           // 128 workgroups per fused pass
  const dim3 gsplit((ND + 255) / 256);

  split_bf16<<<gsplit, 256, 0, stream>>>(f1, f1h, f1l, (int)ND);
  split_bf16<<<gsplit, 256, 0, stream>>>(f2, f2h, f2l, (int)ND);

  // --- stage 1: d2 -> d1 ---
  fused_sim_acc<<<gblk, NTHREADS, 0, stream>>>(f1h, f1l, f2h, f2l, acc1, rs1, N, 1);
  fused_sim_acc<<<gblk, NTHREADS, 0, stream>>>(f1h, f1l, f1h, f1l, two1, rs1, N, 0);
  l1norm_rows<<<N, DF, 0, stream>>>(two1, user_d1);
  gate_rows<<<N, DF, 0, stream>>>(acc1, rs1, f2, Wd1, bd1, Ws1, bs1,
                                  g1W1, g1b1, g1W2, g1b2, user_d1, f1n, N);
  split_bf16<<<gsplit, 256, 0, stream>>>(f1n, f1nh, f1nl, (int)ND);

  // --- stage 2: d1 -> d2 ---
  fused_sim_acc<<<gblk, NTHREADS, 0, stream>>>(f2h, f2l, f1nh, f1nl, acc2, rs2, N, 1);
  fused_sim_acc<<<gblk, NTHREADS, 0, stream>>>(f2h, f2l, f2h, f2l, two2, rs2, N, 0);
  l1norm_rows<<<N, DF, 0, stream>>>(two2, nrm2);
  gate_rows<<<N, DF, 0, stream>>>(acc2, rs2, f1n, Wd2, bd2, Ws2, bs2,
                                  g2W1, g2b1, g2W2, g2b2, nrm2, user_d2, N);
}
